// FastSelfAttention2d_88390426952466
// MI455X (gfx1250) — matrix operand order repaired and checked by reading, not compiled, not run on hardware
//
#include <hip/hip_runtime.h>
#include <hip/hip_bf16.h>

// ---------------------------------------------------------------------------
// FAVOR+ linear self-attention for MI455X (gfx1250, wave32, WMMA).
// All GEMM stages run on v_wmma_f32_16x16x32_f16; intermediates stored f16.
// Each wave now computes 4-5 output tiles per A-fragment (16x64 / 16x80)
// to raise arithmetic intensity; kv stage is split-K with f32 atomics.
// ---------------------------------------------------------------------------

typedef __attribute__((ext_vector_type(16))) _Float16 v16h;
typedef __attribute__((ext_vector_type(8)))  float    v8f;

#define NTOK   16384      // E * L
#define DMODEL 768
#define D3     2304
#define NHEAD  12
#define DHEAD  64
#define MFEAT  256
#define DKV    80         // DHEAD + 16 (col 64 = ones column -> den; 65..79 = 0)
#define KVSPLIT 8

#define WMMA_F16(a, b, c) \
  __builtin_amdgcn_wmma_f32_16x16x32_f16(false, (a), false, (b), (short)0, (c), false, false)

// ---- Fragment loaders (contiguous k = 16*(l/16) + s, same for A and B) ----------------------
// A 16x32 f16: lane l covers row (l&15); K = kb..kb+15,
// kb = (l&16)?16:0.
__device__ __forceinline__ v16h frag_a_f32(const float* __restrict__ A,
                                           int lda, int row0, int k0) {
  int l = threadIdx.x & 31;
  const float* p = A + (long)(row0 + (l & 15)) * lda + k0 + ((l & 16) ? 16 : 0);
  v16h f;
#pragma unroll
  for (int i = 0; i < 8; ++i) f[i] = (_Float16)p[i];
#pragma unroll
  for (int i = 0; i < 8; ++i) f[8 + i] = (_Float16)p[8 + i];
  return f;
}

__device__ __forceinline__ v16h frag_a_f16(const _Float16* __restrict__ A,
                                           long lda, long row0, int k0) {
  int l = threadIdx.x & 31;
  const _Float16* p = A + (row0 + (l & 15)) * lda + k0 + ((l & 16) ? 16 : 0);
  v16h f;
#pragma unroll
  for (int i = 0; i < 8; ++i) f[i] = p[i];
#pragma unroll
  for (int i = 0; i < 8; ++i) f[8 + i] = p[8 + i];
  return f;
}

// B 32x16 from a transposed source BT[col][k] (contiguous in k):
// lane l covers col (l&15); K = kb..kb+15, kb = (l&16)?16:0.
__device__ __forceinline__ v16h frag_bT_f32(const float* __restrict__ BT,
                                            int ldbt, int col0, int k0,
                                            float scale) {
  int l = threadIdx.x & 31;
  const float* p = BT + (long)(col0 + (l & 15)) * ldbt + k0 + ((l & 16) ? 16 : 0);
  v16h f;
#pragma unroll
  for (int i = 0; i < 16; ++i) f[i] = (_Float16)(p[i] * scale);
  return f;
}

// C/D 16x16 f32 layout: VGPR j, lane l -> row = j + ((l&16)?8:0), col = l&15.

// ---------------------------------------------------------------------------
// K1: qkv[N,2304] (f16) = x[N,768] @ W_in^T + b_in
// Wave tile: 16 rows x 64 cols (4 accumulators share one A fragment).
// Block: 4 waves -> 16 x 256 cols. Grid: (1024, 9).
// ---------------------------------------------------------------------------
__global__ __launch_bounds__(128) void qkv_gemm(const float* __restrict__ x,
                                                const float* __restrict__ W,
                                                const float* __restrict__ bias,
                                                _Float16* __restrict__ qkv) {
  int wave = threadIdx.x >> 5;
  int row0 = blockIdx.x * 16;
  int colb = (blockIdx.y * 4 + wave) * 64;
  v8f c[4] = {{}, {}, {}, {}};
  for (int k0 = 0; k0 < DMODEL; k0 += 32) {
    if (k0 + 32 < DMODEL)
      __builtin_prefetch(x + (long)(row0 + (threadIdx.x & 15)) * DMODEL + k0 + 32, 0, 0);
    v16h a = frag_a_f32(x, DMODEL, row0, k0);
#pragma unroll
    for (int t = 0; t < 4; ++t) {
      v16h b = frag_bT_f32(W, DMODEL, colb + 16 * t, k0, 1.0f);
      c[t] = WMMA_F16(a, b, c[t]);
    }
  }
  int l = threadIdx.x & 31;
  int rbase = row0 + ((l & 16) ? 8 : 0);
#pragma unroll
  for (int t = 0; t < 4; ++t) {
    int col = colb + 16 * t + (l & 15);
    float bv = bias[col];
#pragma unroll
    for (int j = 0; j < 8; ++j)
      qkv[(long)(rbase + j) * D3 + col] = (_Float16)(c[t][j] + bv);
  }
}

// ---------------------------------------------------------------------------
// K2: per-(head, token) norms  ||t||^2 / (2*sqrt(DH)) for q and k
// ---------------------------------------------------------------------------
__global__ void norms_kernel(const _Float16* __restrict__ qkv,
                             float* __restrict__ nq, float* __restrict__ nk) {
  long t = (long)blockIdx.x * blockDim.x + threadIdx.x;
  if (t >= (long)NHEAD * NTOK) return;
  int h = (int)(t / NTOK), n = (int)(t % NTOK);
  const _Float16* q = qkv + (long)n * D3 + h * DHEAD;
  const _Float16* k = q + DMODEL;
  float sq = 0.f, sk = 0.f;
#pragma unroll 8
  for (int i = 0; i < DHEAD; ++i) {
    float a = (float)q[i]; sq += a * a;
    float b = (float)k[i]; sk += b * b;
  }
  nq[t] = sq * (1.0f / 16.0f);   // 2*sqrt(64) = 16
  nk[t] = sk * (1.0f / 16.0f);
}

// ---------------------------------------------------------------------------
// K3: feature map  phi = (exp(t @ orf^T / DH^0.25 + mask - norm) + eps)/sqrt(M)
// Wave tile: 16 tokens x 64 features (4 accumulators / A fragment).
// Block: 4 waves -> one 16-token tile covers all 256 features. Grid: (12,1024).
// ---------------------------------------------------------------------------
__global__ __launch_bounds__(128) void featuremap_gemm(
    const _Float16* __restrict__ qkv, const float* __restrict__ orf,
    const float* __restrict__ norms, const int* __restrict__ mask,
    _Float16* __restrict__ outp, int qk_off, int use_mask) {
  int h = blockIdx.x;
  int n0 = blockIdx.y * 16;
  int wave = threadIdx.x >> 5;
  int mb = wave * 64;                             // feature base for this wave
  const _Float16* A = qkv + qk_off + h * DHEAD;   // row n at A + n*D3
  v8f c[4] = {{}, {}, {}, {}};
  for (int k0 = 0; k0 < DHEAD; k0 += 32) {
    v16h a = frag_a_f16(A, D3, n0, k0);
#pragma unroll
    for (int t = 0; t < 4; ++t) {
      v16h b = frag_bT_f32(orf, DHEAD, mb + 16 * t, k0, 0.35355339059f); // 64^-0.25
      c[t] = WMMA_F16(a, b, c[t]);
    }
  }
  int l = threadIdx.x & 31;
  int rbase = n0 + ((l & 16) ? 8 : 0);
#pragma unroll
  for (int t = 0; t < 4; ++t) {
    int m = mb + 16 * t + (l & 15);
#pragma unroll
    for (int j = 0; j < 8; ++j) {
      int n = rbase + j;
      float mt = (use_mask && mask[n]) ? -10000.0f : 0.0f;
      float v = __expf(c[t][j] + mt - norms[(long)h * NTOK + n]);
      outp[((long)h * NTOK + n) * MFEAT + m] = (_Float16)((v + 1e-6f) * 0.0625f);
    }
  }
}

// ---------------------------------------------------------------------------
// K4a: zero kvT accumulator
// ---------------------------------------------------------------------------
__global__ void zero_f32(float* __restrict__ p, int n) {
  int i = blockIdx.x * blockDim.x + threadIdx.x;
  if (i < n) p[i] = 0.f;
}

// ---------------------------------------------------------------------------
// K4b: kvT[h][d][m] += sum_{n in chunk} kp[h][n][m] * vext[n][d]
// (d==64 column -> ksum).  Split-K over 8 chunks with f32 atomics.
// One wave per (h, m-tile, d-tile, k-chunk).  Grid: (12, 16, 5*8).
// ---------------------------------------------------------------------------
__global__ __launch_bounds__(32) void kv_gemm(const _Float16* __restrict__ kp,
                                              const _Float16* __restrict__ qkv,
                                              float* __restrict__ kvT) {
  int h = blockIdx.x;
  int m0 = blockIdx.y * 16;            // 16 tiles over M=256
  int d0 = (blockIdx.z % 5) * 16;      // 5 tiles over DKV=80
  int kbeg = (blockIdx.z / 5) * (NTOK / KVSPLIT);
  int kend = kbeg + (NTOK / KVSPLIT);
  int l = threadIdx.x & 31;
  const _Float16* kph = kp + (long)h * NTOK * MFEAT;
  const _Float16* vh  = qkv + 2 * DMODEL + h * DHEAD;
  int am  = m0 + (l & 15);
  int akb = (l & 16) ? 16 : 0;
  int bd  = d0 + (l & 15);
  int bkb = (l & 16) ? 16 : 0;
  v8f c = {};
  for (int k0 = kbeg; k0 < kend; k0 += 32) {
    v16h a, b;
#pragma unroll
    for (int i = 0; i < 8; ++i) a[i]     = kph[(long)(k0 + akb + i) * MFEAT + am];
#pragma unroll
    for (int i = 0; i < 8; ++i) a[8 + i] = kph[(long)(k0 + akb + 8 + i) * MFEAT + am];
    if (bd < DHEAD) {
#pragma unroll
      for (int i = 0; i < 16; ++i) b[i] = vh[(long)(k0 + bkb + i) * D3 + bd];
    } else {
      _Float16 fv = (bd == DHEAD) ? (_Float16)1.0f : (_Float16)0.0f;
#pragma unroll
      for (int i = 0; i < 16; ++i) b[i] = fv;
    }
    c = WMMA_F16(a, b, c);
  }
  int col = d0 + (l & 15);
  int rbase = m0 + ((l & 16) ? 8 : 0);
#pragma unroll
  for (int j = 0; j < 8; ++j)
    atomicAdd(&kvT[((long)h * DKV + col) * MFEAT + (rbase + j)], c[j]);
}

// ---------------------------------------------------------------------------
// K5: numden[h][n][0..79] = qp[h][n][:] @ kv[h][:][0..79]  (col 64 = den)
// Wave tile: 16 rows x 80 cols (5 accumulators / A fragment). Grid: (12,128).
// ---------------------------------------------------------------------------
__global__ __launch_bounds__(256) void num_gemm(const _Float16* __restrict__ qp,
                                                const float* __restrict__ kvT,
                                                float* __restrict__ numden) {
  int h = blockIdx.x;
  int wave = threadIdx.x >> 5;
  int n0 = (blockIdx.y * 8 + wave) * 16;
  const _Float16* A = qp + (long)h * NTOK * MFEAT;
  const float* BT = kvT + (long)h * DKV * MFEAT;
  v8f c[5] = {{}, {}, {}, {}, {}};
  for (int k0 = 0; k0 < MFEAT; k0 += 32) {
    if (k0 + 32 < MFEAT)
      __builtin_prefetch(A + (long)(n0 + (threadIdx.x & 15)) * MFEAT + k0 + 32, 0, 0);
    v16h a = frag_a_f16(A, MFEAT, n0, k0);
#pragma unroll
    for (int t = 0; t < 5; ++t) {
      v16h b = frag_bT_f32(BT, MFEAT, 16 * t, k0, 1.0f);
      c[t] = WMMA_F16(a, b, c[t]);
    }
  }
  int l = threadIdx.x & 31;
  int rbase = n0 + ((l & 16) ? 8 : 0);
#pragma unroll
  for (int t = 0; t < 5; ++t) {
    int col = 16 * t + (l & 15);
#pragma unroll
    for (int j = 0; j < 8; ++j)
      numden[((long)h * NTOK + rbase + j) * DKV + col] = c[t][j];
  }
}

// ---------------------------------------------------------------------------
// K6: out = LayerNorm(x + num/den) * ln_w + ln_b   (one block per token)
// ---------------------------------------------------------------------------
__global__ __launch_bounds__(256) void final_ln(const float* __restrict__ x,
                                                const float* __restrict__ numden,
                                                const float* __restrict__ lnw,
                                                const float* __restrict__ lnb,
                                                float* __restrict__ out) {
  __shared__ float sbuf[256];
  int n = blockIdx.x;
  float v[3];
  float lsum = 0.f;
#pragma unroll
  for (int i = 0; i < 3; ++i) {
    int d = threadIdx.x + i * 256;
    int h = d >> 6, dh = d & 63;
    const float* nd = numden + ((long)h * NTOK + n) * DKV;
    float val = x[(long)n * DMODEL + d] + nd[dh] / nd[DHEAD];
    v[i] = val;
    lsum += val;
  }
  sbuf[threadIdx.x] = lsum;
  __syncthreads();
  for (int s = 128; s > 0; s >>= 1) {
    if (threadIdx.x < s) sbuf[threadIdx.x] += sbuf[threadIdx.x + s];
    __syncthreads();
  }
  float mean = sbuf[0] * (1.0f / DMODEL);
  __syncthreads();
  float lvar = 0.f;
#pragma unroll
  for (int i = 0; i < 3; ++i) { float dv = v[i] - mean; lvar += dv * dv; }
  sbuf[threadIdx.x] = lvar;
  __syncthreads();
  for (int s = 128; s > 0; s >>= 1) {
    if (threadIdx.x < s) sbuf[threadIdx.x] += sbuf[threadIdx.x + s];
    __syncthreads();
  }
  float inv = rsqrtf(sbuf[0] * (1.0f / DMODEL) + 1e-5f);
#pragma unroll
  for (int i = 0; i < 3; ++i) {
    int d = threadIdx.x + i * 256;
    out[(long)n * DMODEL + d] = (v[i] - mean) * inv * lnw[d] + lnb[d];
  }
}

// ---------------------------------------------------------------------------
extern "C" void kernel_launch(void* const* d_in, const int* in_sizes, int n_in,
                              void* d_out, int out_size, void* d_ws, size_t ws_size,
                              hipStream_t stream) {
  const float* x    = (const float*)d_in[0];
  const float* W    = (const float*)d_in[1];
  const float* bin  = (const float*)d_in[2];
  const float* orf  = (const float*)d_in[3];
  const float* lnw  = (const float*)d_in[4];
  const float* lnb  = (const float*)d_in[5];
  const int*   mask = (const int*)d_in[6];
  float* out = (float*)d_out;

  char* ws = (char*)d_ws;
  size_t off = 0;
  _Float16* qkv_h = (_Float16*)(ws + off); off += (size_t)NTOK * D3 * 2;                 // 75.5 MB
  float*    nq    = (float*)(ws + off);    off += (size_t)NHEAD * NTOK * 4;              // 0.8 MB
  float*    nk    = (float*)(ws + off);    off += (size_t)NHEAD * NTOK * 4;              // 0.8 MB
  _Float16* qp    = (_Float16*)(ws + off); off += (size_t)NHEAD * NTOK * MFEAT * 2;      // 100.7 MB
  _Float16* kp    = (_Float16*)(ws + off); off += (size_t)NHEAD * NTOK * MFEAT * 2;      // 100.7 MB
  float*    kvT   = (float*)(ws + off);    off += (size_t)NHEAD * DKV * MFEAT * 4;       // 1.0 MB
  float*    numden= (float*)(ws + off);    off += (size_t)NHEAD * NTOK * DKV * 4;        // 62.9 MB

  qkv_gemm<<<dim3(NTOK / 16, D3 / 256), 128, 0, stream>>>(x, W, bin, qkv_h);
  norms_kernel<<<(NHEAD * NTOK + 255) / 256, 256, 0, stream>>>(qkv_h, nq, nk);
  featuremap_gemm<<<dim3(NHEAD, NTOK / 16), 128, 0, stream>>>(
      qkv_h, orf, nq, mask, qp, /*qk_off=*/0, /*use_mask=*/0);
  featuremap_gemm<<<dim3(NHEAD, NTOK / 16), 128, 0, stream>>>(
      qkv_h, orf, nk, mask, kp, /*qk_off=*/DMODEL, /*use_mask=*/1);
  int nkvt = NHEAD * DKV * MFEAT;
  zero_f32<<<(nkvt + 255) / 256, 256, 0, stream>>>(kvT, nkvt);
  kv_gemm<<<dim3(NHEAD, MFEAT / 16, (DKV / 16) * KVSPLIT), 32, 0, stream>>>(kp, qkv_h, kvT);
  num_gemm<<<dim3(NHEAD, NTOK / (16 * 8)), 256, 0, stream>>>(qp, kvT, numden);
  final_ln<<<NTOK, 256, 0, stream>>>(x, numden, lnw, lnb, out);
}
